// HingebasedCrossAttentionBLIP2_51367808860581
// MI455X (gfx1250) — matrix operand order repaired and checked by reading, not compiled, not run on hardware
//
#include <hip/hip_runtime.h>

// Problem constants (from reference): B=128, L=256, D=1024
#define DIM  1024
#define BSZ  128
#define LSEQ 256

typedef __attribute__((ext_vector_type(16))) __bf16          v16bf;
typedef __attribute__((ext_vector_type(16))) unsigned short  v16us;
typedef __attribute__((ext_vector_type(8)))  float           v8f;

// Split an fp32 value into bf16 "hi" (truncated) and bf16 "lo" (remainder).
// hi + lo reconstructs x to ~2^-17 relative accuracy -> 3-WMMA products give
// near-fp32 GEMM accuracy on the bf16 matrix pipe.
__device__ __forceinline__ void split_bf16(float f, unsigned short& h, unsigned short& l) {
  unsigned u = __float_as_uint(f);
  float hi_f = __uint_as_float(u & 0xFFFF0000u);
  h = (unsigned short)(u >> 16);
  l = (unsigned short)(__float_as_uint(f - hi_f) >> 16);
}

// Out[m,n] = sum_e A[m,e] * (TRANSW ? W[e,n] : W[n,e]) + rowScale[m]*bias[n]
// A: M x 1024 rows with arbitrary row stride lda. M=128, N=K=1024.
// One wave computes one 16x16 tile via v_wmma_f32_16x16x32_bf16 (hi/lo split).
template<bool TRANSW>
__global__ __launch_bounds__(256) void wmma_linear_kernel(
    const float* __restrict__ A, long lda,
    const float* __restrict__ W,
    const float* __restrict__ bias,      // nullable
    const float* __restrict__ rowScale,  // nullable (coefficient on bias per row)
    float* __restrict__ Out)
{
  const int wave     = threadIdx.x >> 5;
  const int lane     = threadIdx.x & 31;
  const int tile     = blockIdx.x * 8 + wave;      // 512 tiles total (8 M x 64 N)
  const int m0       = (tile & 7) << 4;
  const int n0       = (tile >> 3) << 4;
  const int laneHalf = lane >> 4;                  // 0 or 1
  const int lane15   = lane & 15;

  const int am = m0 + lane15;                      // A fragment row (M)
  const int bn = n0 + lane15;                      // B fragment col (N)

  // Per-wave base pointers hoisted out of the K loop.
  const float* __restrict__ aRow = A + (long)am * lda;   // A row for this lane
  const float* __restrict__ bRow = W + (long)bn * DIM;   // W row (non-trans: k contiguous)
  const float* __restrict__ bCol = W + bn;               // W col (trans: stride DIM)

  v8f acc = {0.f, 0.f, 0.f, 0.f, 0.f, 0.f, 0.f, 0.f};

  for (int kk = 0; kk < DIM; kk += 32) {
    v16us ah, al, bh, bl;
    // A-matrix 16x32 bf16 layout (ISA 7.12.2):
    // lanes0-15: K=0..7 (VGPR0-3), 16..23 (VGPR4-7); lanes16-31: K=8..15, 24..31
#pragma unroll
    for (int i = 0; i < 8; ++i) {
      int k = kk + ((i >> 2) << 4) + (laneHalf << 3) + ((i & 3) << 1);
      float2 a = *(const float2*)(aRow + k);             // global_load_b64
      unsigned short h0, l0, h1, l1;
      split_bf16(a.x, h0, l0);
      split_bf16(a.y, h1, l1);
      ah[2 * i] = h0; ah[2 * i + 1] = h1;
      al[2 * i] = l0; al[2 * i + 1] = l1;
    }
    // B-matrix 32x16 bf16 layout (same K order as A):
    // lanes0-15: K=0..7, 16..23; lanes16-31: K=8..15, 24..31 (low half = even K).
#pragma unroll
    for (int i = 0; i < 8; ++i) {
      int k = kk + ((i >> 2) << 4) + (laneHalf << 3) + ((i & 3) << 1);
      float b0, b1;
      if (TRANSW) {
        b0 = bCol[(long)k * DIM];
        b1 = bCol[(long)(k + 1) * DIM];
      } else {
        float2 b = *(const float2*)(bRow + k);           // global_load_b64
        b0 = b.x; b1 = b.y;
      }
      unsigned short h0, l0, h1, l1;
      split_bf16(b0, h0, l0);
      split_bf16(b1, h1, l1);
      bh[2 * i] = h0; bh[2 * i + 1] = h1;
      bl[2 * i] = l0; bl[2 * i + 1] = l1;
    }
    v16bf ahv = __builtin_bit_cast(v16bf, ah);
    v16bf alv = __builtin_bit_cast(v16bf, al);
    v16bf bhv = __builtin_bit_cast(v16bf, bh);
    v16bf blv = __builtin_bit_cast(v16bf, bl);
    acc = __builtin_amdgcn_wmma_f32_16x16x32_bf16(false, ahv, false, bhv, (short)0, acc, false, false);
    acc = __builtin_amdgcn_wmma_f32_16x16x32_bf16(false, ahv, false, blv, (short)0, acc, false, false);
    acc = __builtin_amdgcn_wmma_f32_16x16x32_bf16(false, alv, false, bhv, (short)0, acc, false, false);
  }

  float bval = bias ? bias[bn] : 0.f;
  // D-matrix 16x16 f32 layout: VGPR v -> M = v + 8*laneHalf, N = lane15
  float* __restrict__ oCol = Out + (long)(m0 + (laneHalf << 3)) * DIM + bn;
#pragma unroll
  for (int v = 0; v < 8; ++v) {
    int m = m0 + v + (laneHalf << 3);
    float rs = rowScale ? rowScale[m] : 1.f;
    oCol[(long)v * DIM] = acc[v] + rs * bval;
  }
}

// out[row] = scale * ( X[row,:] . V[b,:] + rowAdd[b] ),  b = row >> rowsPerBShift
// One wave per 1024-long row; float4 streaming loads (global_load_b128).
__global__ __launch_bounds__(256) void rowdot_kernel(
    const float* __restrict__ X,
    const float* __restrict__ V,
    const float* __restrict__ rowAdd,  // nullable
    float* __restrict__ out,
    int vStride, int rowsPerBShift, float scale, int nRows)
{
  int row = (int)((blockIdx.x * blockDim.x + threadIdx.x) >> 5);
  if (row >= nRows) return;
  int lane = threadIdx.x & 31;
  int b = row >> rowsPerBShift;
  const float4* xr = (const float4*)(X + (long)row * DIM);
  const float4* vr = (const float4*)(V + (long)b * vStride);
  float s = 0.f;
#pragma unroll
  for (int w = 0; w < 8; ++w) {
    float4 x = xr[w * 32 + lane];
    float4 v = vr[w * 32 + lane];
    s += x.x * v.x + x.y * v.y + x.z * v.z + x.w * v.w;
  }
#pragma unroll
  for (int off = 16; off > 0; off >>= 1) s += __shfl_down(s, off, 32);
  if (lane == 0) {
    float add = rowAdd ? rowAdd[b] : 0.f;
    out[row] = scale * (s + add);
  }
}

// out[b,e] = sum_t wts[b,t] * X[b,t,e] ; optionally Ssum[b] = sum_t wts[b,t].
// One block per batch b; threads cover e in float4 chunks; wts staged in LDS.
__global__ __launch_bounds__(256) void weighted_accum_kernel(
    const float* __restrict__ X,    // [B, LSEQ, DIM]
    const float* __restrict__ wts,  // [B, LSEQ]
    float* __restrict__ out,        // [B, DIM]
    float* __restrict__ Ssum)       // [B] or nullptr
{
  __shared__ float sw[LSEQ];
  __shared__ float red[LSEQ];
  int b = blockIdx.x, t = threadIdx.x;
  float w = wts[b * LSEQ + t];
  sw[t] = w;
  red[t] = w;
  __syncthreads();
  for (int off = 128; off > 0; off >>= 1) {
    if (t < off) red[t] += red[t + off];
    __syncthreads();
  }
  if (t == 0 && Ssum) Ssum[b] = red[0];

  const float4* xb = (const float4*)(X + (long)b * LSEQ * DIM);
  float4 acc = {0.f, 0.f, 0.f, 0.f};
#pragma unroll 4
  for (int c = 0; c < LSEQ; ++c) {
    float4 x = xb[c * (DIM / 4) + t];
    float wc = sw[c];
    acc.x += wc * x.x; acc.y += wc * x.y; acc.z += wc * x.z; acc.w += wc * x.w;
  }
  ((float4*)(out + (long)b * DIM))[t] = acc;
}

// Row-wise softmax over LSEQ=256, one block per row.
__global__ __launch_bounds__(256) void softmax_kernel(
    const float* __restrict__ in, float* __restrict__ out)
{
  __shared__ float red[LSEQ];
  int b = blockIdx.x, t = threadIdx.x;
  float v = in[b * LSEQ + t];
  red[t] = v;
  __syncthreads();
  for (int off = 128; off > 0; off >>= 1) {
    if (t < off) red[t] = fmaxf(red[t], red[t + off]);
    __syncthreads();
  }
  float mx = red[0];
  __syncthreads();
  float e = expf(v - mx);
  red[t] = e;
  __syncthreads();
  for (int off = 128; off > 0; off >>= 1) {
    if (t < off) red[t] += red[t + off];
    __syncthreads();
  }
  out[b * LSEQ + t] = e / red[0];
}

extern "C" void kernel_launch(void* const* d_in, const int* in_sizes, int n_in,
                              void* d_out, int out_size, void* d_ws, size_t ws_size,
                              hipStream_t stream) {
  const float* ref    = (const float*)d_in[0];   // [128,256,1024]
  const float* cap    = (const float*)d_in[1];   // [128,256,1024]
  const float* tar    = (const float*)d_in[2];   // [128,256,1024]
  const float* qr_w   = (const float*)d_in[3];
  const float* qr_b   = (const float*)d_in[4];
  const float* ktxt_w = (const float*)d_in[5];
  const float* ktxt_b = (const float*)d_in[6];
  const float* ktar_w = (const float*)d_in[7];
  const float* ktar_b = (const float*)d_in[8];
  const float* v_w    = (const float*)d_in[9];
  const float* v_b    = (const float*)d_in[10];
  float* out = (float*)d_out;                    // [128,1024]
  float* ws  = (float*)d_ws;

  // Workspace layout (floats): six 128x1024 mats, three 128x256 mats, three 128-vecs
  float* Q0     = ws + 0;        // [B,D] q_ref row0
  float* U      = ws + 131072;   // [B,D] ktxt_w^T q0
  float* G      = ws + 262144;   // [B,D] caption^T attA0
  float* Wv     = ws + 393216;   // [B,D] w
  float* H      = ws + 524288;   // [B,D] ktar_w^T w
  float* P      = ws + 655360;   // [B,D] target^T attC
  float* attA0  = ws + 786432;   // [B,L]
  float* scores = ws + 819200;   // [B,L]
  float* attC   = ws + 851968;   // [B,L]
  float* C0     = ws + 884736;   // [B] q0 . ktxt_b
  float* D0     = ws + 884864;   // [B] w . ktar_b
  float* Ssum   = ws + 884992;   // [B] sum_c attA0

  const float scale = 0.03125f;  // 1/sqrt(1024)

  // 1) Q0 = ref[:,0,:] @ qr_w^T + qr_b   (A row stride = L*D picks row 0 per batch)
  wmma_linear_kernel<false><<<64, 256, 0, stream>>>(ref, (long)LSEQ * DIM, qr_w, qr_b, nullptr, Q0);
  // 2) C0[b] = Q0[b] . ktxt_b
  rowdot_kernel<<<16, 256, 0, stream>>>(Q0, ktxt_b, nullptr, C0, 0, 0, 1.0f, BSZ);
  // 3) U = Q0 @ ktxt_w
  wmma_linear_kernel<true><<<64, 256, 0, stream>>>(Q0, (long)DIM, ktxt_w, nullptr, nullptr, U);
  // 4) attA0[b,c] = scale*(caption[b,c] . U[b] + C0[b])      -- streams caption (128 MB)
  rowdot_kernel<<<4096, 256, 0, stream>>>(cap, U, C0, attA0, DIM, 8, scale, BSZ * LSEQ);
  // 5) G[b] = caption[b]^T attA0[b] ; Ssum[b] = sum attA0[b]
  weighted_accum_kernel<<<128, 256, 0, stream>>>(cap, attA0, G, Ssum);
  // 6) Wv = G @ ktxt_w^T + Ssum * ktxt_b
  wmma_linear_kernel<false><<<64, 256, 0, stream>>>(G, (long)DIM, ktxt_w, ktxt_b, Ssum, Wv);
  // 7) D0[b] = Wv[b] . ktar_b
  rowdot_kernel<<<16, 256, 0, stream>>>(Wv, ktar_b, nullptr, D0, 0, 0, 1.0f, BSZ);
  // 8) H = Wv @ ktar_w
  wmma_linear_kernel<true><<<64, 256, 0, stream>>>(Wv, (long)DIM, ktar_w, nullptr, nullptr, H);
  // 9) scores[b,t] = scale*(target[b,t] . H[b] + D0[b])      -- streams target (128 MB)
  rowdot_kernel<<<4096, 256, 0, stream>>>(tar, H, D0, scores, DIM, 8, scale, BSZ * LSEQ);
  // 10) attC = softmax(scores)
  softmax_kernel<<<128, 256, 0, stream>>>(scores, attC);
  // 11) P[b] = target[b]^T attC[b]
  weighted_accum_kernel<<<128, 256, 0, stream>>>(tar, attC, P, nullptr);
  // 12) out = P @ v_w^T + v_b   (softmax row-sum == 1 -> plain bias)
  wmma_linear_kernel<false><<<64, 256, 0, stream>>>(P, (long)DIM, v_w, v_b, nullptr, out);
}
